// MambaBlock_45655502356848
// MI455X (gfx1250) — matrix operand order repaired and checked by reading, not compiled, not run on hardware
//
#include <hip/hip_runtime.h>
#include <hip/hip_bf16.h>

// ---------------- dimensions (fixed by reference) ----------------
#define BATCH   2
#define SEQ     4096
#define DMODEL  1024
#define DINNER  2048
#define HEADDIM 64
#define NHEADS  32
#define DSTATE  128
#define DCONV   4
#define CHUNK   256
#define NCHUNK  (SEQ / CHUNK)           // 16
#define DINPROJ 4384                    // 2*2048 + 2*128 + 32
#define CONVDIM 2304                    // 2048 + 2*128
#define ROWS    (BATCH * SEQ)           // 8192
#define KPANEL  256                     // K staging panel for TDM GEMM

typedef __bf16 bf16_t;
typedef __attribute__((ext_vector_type(16))) __bf16 v16bf;
typedef __attribute__((ext_vector_type(8)))  float  v8f;
typedef unsigned int u32x4 __attribute__((ext_vector_type(4)));
typedef int          i32x4 __attribute__((ext_vector_type(4)));
typedef int          i32x8 __attribute__((ext_vector_type(8)));

// ---------------- WMMA helpers (gfx1250 wave32 layouts) ----------------
__device__ __forceinline__ v8f wmma_bf16(v16bf a, v16bf b, v8f c) {
  return __builtin_amdgcn_wmma_f32_16x16x32_bf16(
      /*neg_a=*/false, a, /*neg_b=*/false, b,
      /*c_mod=*/(short)0, c, /*reuse_a=*/false, /*reuse_b=*/false);
}

// A fragment K index for vgpr j (16-bit A 16x32 layout)
__device__ __forceinline__ int a_k_of(int lane, int j) {
  return ((j >> 2) << 4) + ((lane >> 4) << 3) + ((j & 3) << 1);
}
// B fragment K index for vgpr j (16-bit B 32x16 layout)
__device__ __forceinline__ int b_k_of(int lane, int j) {
  return ((j >> 2) << 4) + ((lane >> 4) << 3) + ((j & 3) << 1);
}

// A: row-major M x K tile at (m0,k0)
__device__ __forceinline__ v16bf load_a(const bf16_t* A, int ld, int m0, int k0) {
  int lane = threadIdx.x & 31;
  const bf16_t* row = A + (size_t)(m0 + (lane & 15)) * ld + k0;
  v16bf f;
#pragma unroll
  for (int j = 0; j < 8; ++j) {
    int k = a_k_of(lane, j);
    f[2 * j] = row[k];
    f[2 * j + 1] = row[k + 1];
  }
  return f;
}

// A row-major, each column k scaled by sk[k0+k] (LDS decay vector)
__device__ __forceinline__ v16bf load_a_kscale(const bf16_t* A, int ld, int m0,
                                               int k0, const float* sk) {
  int lane = threadIdx.x & 31;
  const bf16_t* row = A + (size_t)(m0 + (lane & 15)) * ld + k0;
  v16bf f;
#pragma unroll
  for (int j = 0; j < 8; ++j) {
    int k = a_k_of(lane, j);
    f[2 * j]     = (bf16_t)((float)row[k] * sk[k0 + k]);
    f[2 * j + 1] = (bf16_t)((float)row[k + 1] * sk[k0 + k + 1]);
  }
  return f;
}

// A row-major, whole row scaled by per-lane scalar sm
__device__ __forceinline__ v16bf load_a_mscale(const bf16_t* A, int ld, int m0,
                                               int k0, float sm) {
  int lane = threadIdx.x & 31;
  const bf16_t* row = A + (size_t)(m0 + (lane & 15)) * ld + k0;
  v16bf f;
#pragma unroll
  for (int j = 0; j < 8; ++j) {
    int k = a_k_of(lane, j);
    f[2 * j]     = (bf16_t)((float)row[k] * sm);
    f[2 * j + 1] = (bf16_t)((float)row[k + 1] * sm);
  }
  return f;
}

// A tile of M = tril(G * exp(Acs[m]-Acs[s])) built on the fly from f32 G
__device__ __forceinline__ v16bf load_a_M(const float* G, int m0, int k0,
                                          const float* Acs) {
  int lane = threadIdx.x & 31;
  int m = m0 + (lane & 15);
  float am = Acs[m];
  const float* row = G + (size_t)m * CHUNK + k0;
  v16bf f;
#pragma unroll
  for (int j = 0; j < 8; ++j) {
    int k = a_k_of(lane, j);
    int s0 = k0 + k, s1 = s0 + 1;
    float v0 = (s0 <= m) ? row[k] * __expf(am - Acs[s0]) : 0.f;
    float v1 = (s1 <= m) ? row[k + 1] * __expf(am - Acs[s1]) : 0.f;
    f[2 * j] = (bf16_t)v0;
    f[2 * j + 1] = (bf16_t)v1;
  }
  return f;
}

// B fragment from N-major storage: element B[k][n] = Bt[n*ld + k] (contiguous k)
__device__ __forceinline__ v16bf load_bT(const bf16_t* Bt, int ld, int n0, int k0) {
  int lane = threadIdx.x & 31;
  const bf16_t* row = Bt + (size_t)(n0 + (lane & 15)) * ld + k0;
  v16bf f;
#pragma unroll
  for (int j = 0; j < 8; ++j) {
    int k = b_k_of(lane, j);
    f[2 * j] = row[k];
    f[2 * j + 1] = row[k + 1];
  }
  return f;
}

// Same, but from fp32 N-major source with on-the-fly bf16 convert
__device__ __forceinline__ v16bf load_bT_f32(const float* Bt, int ld, int n0, int k0) {
  int lane = threadIdx.x & 31;
  const float* row = Bt + (size_t)(n0 + (lane & 15)) * ld + k0;
  v16bf f;
#pragma unroll
  for (int j = 0; j < 8; ++j) {
    int k = b_k_of(lane, j);
    f[2 * j] = (bf16_t)row[k];
    f[2 * j + 1] = (bf16_t)row[k + 1];
  }
  return f;
}

// C/D 16x16 f32 store
__device__ __forceinline__ void store_c(float* C, int ld, int m0, int n0, v8f acc) {
  int lane = threadIdx.x & 31;
  int n = n0 + (lane & 15);
  int mb = m0 + ((lane >> 4) << 3);
#pragma unroll
  for (int j = 0; j < 8; ++j) C[(size_t)(mb + j) * ld + n] = acc[j];
}

// ---------------- elementwise helpers ----------------
__device__ __forceinline__ float silu_f(float x) { return x / (1.f + __expf(-x)); }

// ---------------- kernels ----------------

__global__ void k_cast_bf16(const float* __restrict__ in, bf16_t* __restrict__ out,
                            size_t n) {
  size_t i = (size_t)blockIdx.x * 256 + threadIdx.x;
  if (i < n) out[i] = (bf16_t)in[i];
}

// in: R x C row-major (f32) -> out: C x R (bf16)
__global__ void k_cast_transpose(const float* __restrict__ in,
                                 bf16_t* __restrict__ out, int R, int C) {
  size_t i = (size_t)blockIdx.x * 256 + threadIdx.x;
  if (i >= (size_t)R * C) return;
  int r = (int)(i / C), c = (int)(i % C);
  out[(size_t)c * R + r] = (bf16_t)in[i];
}

// issue one TDM 2D tile load: 32 rows x KPANEL elems of bf16, row stride K,
// into LDS offset ldsoff (bytes).  Descriptor per CDNA5 ISA ch.8 (D#).
__device__ __forceinline__ void tdm_load_panel(const bf16_t* gsrc, int K,
                                               unsigned ldsoff) {
  unsigned long long gaddr = (unsigned long long)(const void*)gsrc;
  u32x4 g0;
  i32x8 g1;
  i32x4 g2 = {};
  i32x4 g3 = {};
  i32x8 g4 = {};
  g0[0] = 1u;                                          // count=1, user mode
  g0[1] = ldsoff;                                      // lds_addr (bytes)
  g0[2] = (unsigned)(gaddr & 0xffffffffu);             // global_addr lo
  g0[3] = (unsigned)((gaddr >> 32) & 0x1ffffffu) | (2u << 30);  // hi | type=2
  unsigned td0 = (unsigned)K, td1 = 32u;               // tensor dims
  unsigned tl0 = (unsigned)KPANEL, tl1 = 32u;          // tile dims
  unsigned long long s0 = (unsigned long long)K;       // dim0 stride (elems)
  g1[0] = (int)(1u << 16);                             // data_size = 2 bytes
  g1[1] = (int)((td0 & 0xffffu) << 16);                // tensor_dim0 lo16
  g1[2] = (int)(((td0 >> 16) & 0xffffu) | ((td1 & 0xffffu) << 16));
  g1[3] = (int)(((td1 >> 16) & 0xffffu) | ((tl0 & 0xffffu) << 16));
  g1[4] = (int)(tl1 & 0xffffu);                        // tile_dim1 (tile_dim2=0)
  g1[5] = (int)(s0 & 0xffffffffu);                     // stride0 lo
  g1[6] = (int)((s0 >> 32) & 0xffffu);                 // stride0 hi (stride1=0)
  g1[7] = 0;
  __builtin_amdgcn_tensor_load_to_lds(g0, g1, g2, g3, g4, 0);
}

// ---------------------------------------------------------------------------
// Register-blocked bf16 GEMM with double-buffered Tensor-Data-Mover staging
// of the weight panel into LDS.  C[M,N] = A[M,K] * B[K,N], B N-major (Bt[N,K]).
// Block = 256 threads (8 waves); block tile = 512(M) x 32(N);
// per wave: 64(M) x 32(N) = 8 accumulators; K staged in 256-wide panels.
// TDM for panel p+1 runs concurrently with WMMA on panel p; the issuing wave
// waits with s_wait_tensorcnt(1) (in-order TDM completion per wave).
// ---------------------------------------------------------------------------
__global__ void k_gemm_bf16_tdm(const bf16_t* __restrict__ A,
                                const bf16_t* __restrict__ Bt,
                                float* __restrict__ C, int M, int N, int K) {
  __shared__ bf16_t Bs[2][32 * KPANEL];              // 2 x 16 KB weight panels
  int nblk = blockIdx.x * 32;
  int mblk = blockIdx.y * 512;
  int wave = threadIdx.x >> 5;
  int lane = threadIdx.x & 31;
  int m0 = mblk + wave * 64;
  (void)M;

  v8f acc[4][2] = {};

  const bf16_t* wbase = Bt + (size_t)nblk * K;       // 32 weight rows
  int np = K / KPANEL;
  if (wave == 0) tdm_load_panel(wbase, K, 0);        // preload panel 0

  for (int p = 0; p < np; ++p) {
    if (wave == 0) {
      if (p + 1 < np) {
        tdm_load_panel(wbase + (p + 1) * KPANEL, K,
                       (unsigned)(((p + 1) & 1) * 32 * KPANEL * 2));
        __builtin_amdgcn_s_wait_tensorcnt(1);        // panel p landed, p+1 in flight
      } else {
        __builtin_amdgcn_s_wait_tensorcnt(0);
      }
    }
    __syncthreads();

    int kp = p * KPANEL;
    // The TDM writes Bs behind the compiler's back (the builtin carries no
    // pointer).  Escape the LDS address through an asm with a memory clobber
    // so the staged weights are genuinely re-loaded -- but perform the actual
    // fragment loads through the Bs array expression itself so the LDS
    // address space survives and the loads lower to ds_load (DScnt only),
    // decoupled from the global A-stream's LOADcnt.
    {
      const bf16_t* esc = &Bs[0][0];
      asm volatile("" : "+v"(esc) : : "memory");
    }
    const bf16_t* Bp = &Bs[p & 1][0];

    // prefetch next A panel (streamed operand)
    if (p + 1 < np)
      __builtin_prefetch(A + (size_t)(m0 + (lane & 15)) * K + kp + KPANEL, 0, 0);

#pragma unroll 2
    for (int k0 = 0; k0 < KPANEL; k0 += 32) {
      v16bf b0 = load_bT(Bp, KPANEL, 0, k0);
      v16bf b1 = load_bT(Bp, KPANEL, 16, k0);
#pragma unroll
      for (int i = 0; i < 4; ++i) {
        v16bf a = load_a(A, K, m0 + i * 16, kp + k0);
        acc[i][0] = wmma_bf16(a, b0, acc[i][0]);
        acc[i][1] = wmma_bf16(a, b1, acc[i][1]);
      }
    }
    __syncthreads();
  }
#pragma unroll
  for (int i = 0; i < 4; ++i) {
    store_c(C, N, m0 + i * 16, nblk, acc[i][0]);
    store_c(C, N, m0 + i * 16, nblk + 16, acc[i][1]);
  }
}

// dt = softplus(raw + bias); adt = dt * (-exp(A_log))
__global__ void k_dt(const float* __restrict__ zx, const float* __restrict__ dt_bias,
                     const float* __restrict__ A_log, float* __restrict__ dtv,
                     float* __restrict__ adt) {
  int idx = blockIdx.x * 256 + threadIdx.x;        // over ROWS*NHEADS
  int h = idx & (NHEADS - 1);
  size_t row = (size_t)(idx >> 5);
  float x = zx[row * DINPROJ + (2 * DINNER + 2 * DSTATE) + h] + dt_bias[h];
  float sp = (x > 20.f) ? x : log1pf(__expf(x));
  dtv[idx] = sp;
  adt[idx] = -sp * __expf(A_log[h]);
}

// causal depthwise conv + silu, then route channels
__global__ void k_conv(const float* __restrict__ zx, const float* __restrict__ conv_w,
                       const float* __restrict__ conv_b, const float* __restrict__ dtv,
                       float* __restrict__ xs, bf16_t* __restrict__ xdtT,
                       bf16_t* __restrict__ Bb, bf16_t* __restrict__ Btb,
                       bf16_t* __restrict__ Cb) {
  int c = blockIdx.x * 256 + threadIdx.x;          // 0..CONVDIM-1
  int l = blockIdx.y;
  int b = blockIdx.z;
  float acc = conv_b[c];
#pragma unroll
  for (int k = 0; k < DCONV; ++k) {
    int t = l - (DCONV - 1) + k;
    if (t >= 0)
      acc += zx[((size_t)b * SEQ + t) * DINPROJ + DINNER + c] * conv_w[c * DCONV + k];
  }
  float v = silu_f(acc);
  size_t bl = (size_t)b * SEQ + l;
  if (c < DINNER) {
    xs[bl * DINNER + c] = v;
    int h = c >> 6, p = c & 63;
    float d = dtv[bl * NHEADS + h];
    xdtT[(((size_t)b * NHEADS + h) * HEADDIM + p) * SEQ + l] = (bf16_t)(v * d);
  } else if (c < DINNER + DSTATE) {
    int n = c - DINNER;
    Bb[bl * DSTATE + n] = (bf16_t)v;
    Btb[((size_t)b * DSTATE + n) * SEQ + l] = (bf16_t)v;
  } else {
    int n = c - DINNER - DSTATE;
    Cb[bl * DSTATE + n] = (bf16_t)v;
  }
}

// per (b,chunk): G[l][s] = sum_n C[l][n]*B[s][n]; wave = 1 mt x 4 st tiles
__global__ void k_gram(const bf16_t* __restrict__ Cb, const bf16_t* __restrict__ Bb,
                       float* __restrict__ G) {
  int bc = blockIdx.z;                              // b*NCHUNK + c
  int st0 = blockIdx.x * 64;                        // 4 tiles of s
  int mt = blockIdx.y;
  const bf16_t* Crow = Cb + (size_t)bc * CHUNK * DSTATE;
  const bf16_t* Brow = Bb + (size_t)bc * CHUNK * DSTATE;
  v8f acc[4] = {};
  for (int k0 = 0; k0 < DSTATE; k0 += 32) {
    v16bf a = load_a(Crow, DSTATE, mt * 16, k0);
#pragma unroll
    for (int i = 0; i < 4; ++i)
      acc[i] = wmma_bf16(a, load_bT(Brow, DSTATE, st0 + i * 16, k0), acc[i]);
  }
  float* Gc = G + (size_t)bc * CHUNK * CHUNK;
#pragma unroll
  for (int i = 0; i < 4; ++i) store_c(Gc, CHUNK, mt * 16, st0 + i * 16, acc[i]);
}

// per (b,c,h): S[p][n] = sum_l X[p][l]*exp(tot-Acs[l])*B[l][n]
// wave = 1 pt x 4 nt tiles -> k-scaled A fragment built once per k-step
__global__ void k_states(const float* __restrict__ adt, const bf16_t* __restrict__ xdtT,
                         const bf16_t* __restrict__ Btb, float* __restrict__ S,
                         float* __restrict__ chunkA) {
  int c = blockIdx.x, h = blockIdx.y, b = blockIdx.z;
  __shared__ float Acs[CHUNK];
  __shared__ float ds[CHUNK];
  int l = threadIdx.x;
  Acs[l] = adt[(((size_t)b * SEQ) + c * CHUNK + l) * NHEADS + h];
  __syncthreads();
  for (int off = 1; off < CHUNK; off <<= 1) {
    float v = Acs[l];
    if (l >= off) v += Acs[l - off];
    __syncthreads();
    Acs[l] = v;
    __syncthreads();
  }
  float tot = Acs[CHUNK - 1];
  ds[l] = __expf(tot - Acs[l]);
  if (l == 0) chunkA[((size_t)b * NHEADS + h) * NCHUNK + c] = tot;
  __syncthreads();

  int wave = threadIdx.x >> 5;
  int pt = wave & 3;                 // 4 p-tiles
  int nb = (wave >> 2) * 64;         // 2 groups x 4 n-tiles
  const bf16_t* X = xdtT + (((size_t)b * NHEADS + h) * HEADDIM) * SEQ + (size_t)c * CHUNK;
  const bf16_t* BT = Btb + ((size_t)b * DSTATE) * SEQ + (size_t)c * CHUNK;
  float* Sout = S + (((size_t)b * NCHUNK + c) * NHEADS + h) * (HEADDIM * DSTATE);
  v8f acc[4] = {};
  for (int k0 = 0; k0 < CHUNK; k0 += 32) {
    v16bf a = load_a_kscale(X, SEQ, pt * 16, k0, ds);
#pragma unroll
    for (int i = 0; i < 4; ++i)
      acc[i] = wmma_bf16(a, load_bT(BT, SEQ, nb + i * 16, k0), acc[i]);
  }
#pragma unroll
  for (int i = 0; i < 4; ++i)
    store_c(Sout, DSTATE, pt * 16, nb + i * 16, acc[i]);
}

// inter-chunk recurrence, in place: S[c] <- R[c], R[c]=exp(sumA[c-1])*R[c-1]+S[c-1]
__global__ void k_scan(float* __restrict__ S, const float* __restrict__ chunkA) {
  int bh = blockIdx.x;                              // b*NHEADS + h
  int b = bh >> 5, h = bh & 31;
  const float* cA = chunkA + (size_t)bh * NCHUNK;
  for (int j = 0; j < (HEADDIM * DSTATE) / 256; ++j) {
    int e = threadIdx.x + j * 256;
    float R = 0.f, prevSt = 0.f, prevA = 0.f;
    for (int c = 0; c < NCHUNK; ++c) {
      size_t idx = (((size_t)b * NCHUNK + c) * NHEADS + h) * (HEADDIM * DSTATE) + e;
      float st = S[idx];
      if (c > 0) R = __expf(prevA) * R + prevSt;
      S[idx] = R;
      prevSt = st;
      prevA = cA[c];
    }
  }
}

// per (b,c,h): y = tril(G*decay)@Xc + (C*out_decay)@S^T + xs*D
// wave handles 2 mt values; all 4 nt accumulated per A fragment build
__global__ void k_y(const float* __restrict__ adt, const float* __restrict__ G,
                    const bf16_t* __restrict__ xdtT, const bf16_t* __restrict__ Cb,
                    const float* __restrict__ S, const float* __restrict__ xs,
                    const float* __restrict__ Dp, float* __restrict__ y) {
  int c = blockIdx.x, h = blockIdx.y, b = blockIdx.z;
  __shared__ float Acs[CHUNK];
  int l = threadIdx.x;
  Acs[l] = adt[(((size_t)b * SEQ) + c * CHUNK + l) * NHEADS + h];
  __syncthreads();
  for (int off = 1; off < CHUNK; off <<= 1) {
    float v = Acs[l];
    if (l >= off) v += Acs[l - off];
    __syncthreads();
    Acs[l] = v;
    __syncthreads();
  }
  int wave = threadIdx.x >> 5, lane = threadIdx.x & 31;
  const float* Gc = G + (size_t)(b * NCHUNK + c) * CHUNK * CHUNK;
  const bf16_t* X = xdtT + (((size_t)b * NHEADS + h) * HEADDIM) * SEQ + (size_t)c * CHUNK;
  const bf16_t* Cc = Cb + (((size_t)b * SEQ) + (size_t)c * CHUNK) * DSTATE;
  const float* Sc = S + (((size_t)b * NCHUNK + c) * NHEADS + h) * (HEADDIM * DSTATE);
  float Dh = Dp[h];
  for (int mt = wave; mt < 16; mt += 8) {
    v8f acc[4] = {};
    // intra-chunk: masked M(256x256) @ Xc(256x64); Xc N-major via xdtT
    for (int k0 = 0; k0 < (mt + 1) * 16; k0 += 32) {
      v16bf a = load_a_M(Gc, mt * 16, k0, Acs);
#pragma unroll
      for (int nt = 0; nt < 4; ++nt)
        acc[nt] = wmma_bf16(a, load_bT(X, SEQ, nt * 16, k0), acc[nt]);
    }
    // inter-chunk: (C*exp(Acs_row))(256x128) @ S^T(128x64)
    float sm = __expf(Acs[mt * 16 + (lane & 15)]);
    for (int k0 = 0; k0 < DSTATE; k0 += 32) {
      v16bf a = load_a_mscale(Cc, DSTATE, mt * 16, k0, sm);
#pragma unroll
      for (int nt = 0; nt < 4; ++nt)
        acc[nt] = wmma_bf16(a, load_bT_f32(Sc, DSTATE, nt * 16, k0), acc[nt]);
    }
    int nloc = lane & 15;
    int mb = mt * 16 + ((lane >> 4) << 3);
#pragma unroll
    for (int nt = 0; nt < 4; ++nt) {
      int n = nt * 16 + nloc;
#pragma unroll
      for (int j = 0; j < 8; ++j) {
        size_t row = (size_t)b * SEQ + c * CHUNK + mb + j;
        size_t col = (size_t)h * HEADDIM + n;
        float xv = xs[row * DINNER + col];
        y[row * DINNER + col] = acc[nt][j] + xv * Dh;
      }
    }
  }
}

// gated RMSNorm -> bf16 u
__global__ void k_rmsnorm(const float* __restrict__ y, const float* __restrict__ zx,
                          const float* __restrict__ norm_w, bf16_t* __restrict__ u) {
  int row = blockIdx.x;
  __shared__ float red[256];
  const float* yr = y + (size_t)row * DINNER;
  const float* zr = zx + (size_t)row * DINPROJ;     // z = first 2048 cols
  float local[DINNER / 256];
  float ss = 0.f;
#pragma unroll
  for (int j = 0; j < DINNER / 256; ++j) {
    int cidx = threadIdx.x + j * 256;
    float v = yr[cidx] * silu_f(zr[cidx]);
    local[j] = v;
    ss += v * v;
  }
  red[threadIdx.x] = ss;
  __syncthreads();
  for (int off = 128; off > 0; off >>= 1) {
    if (threadIdx.x < off) red[threadIdx.x] += red[threadIdx.x + off];
    __syncthreads();
  }
  float scale = rsqrtf(red[0] / (float)DINNER + 1e-5f);
#pragma unroll
  for (int j = 0; j < DINNER / 256; ++j) {
    int cidx = threadIdx.x + j * 256;
    u[(size_t)row * DINNER + cidx] = (bf16_t)(local[j] * scale * norm_w[cidx]);
  }
}

// ---------------- launch ----------------
extern "C" void kernel_launch(void* const* d_in, const int* in_sizes, int n_in,
                              void* d_out, int out_size, void* d_ws, size_t ws_size,
                              hipStream_t stream) {
  const float* x       = (const float*)d_in[0];
  const float* W_in    = (const float*)d_in[1];
  const float* conv_w  = (const float*)d_in[2];
  const float* conv_b  = (const float*)d_in[3];
  const float* dt_bias = (const float*)d_in[4];
  const float* A_log   = (const float*)d_in[5];
  const float* Dp      = (const float*)d_in[6];
  const float* norm_w  = (const float*)d_in[7];
  const float* W_out   = (const float*)d_in[8];
  float* out = (float*)d_out;

  // workspace carve-up (256B aligned)
  char* ws = (char*)d_ws;
  size_t off = 0;
  auto take = [&](size_t bytes) {
    char* p = ws + off;
    off = (off + bytes + 255) & ~(size_t)255;
    return p;
  };
  float*  zx     = (float*)take((size_t)ROWS * DINPROJ * 4);
  bf16_t* x_bf   = (bf16_t*)take((size_t)ROWS * DMODEL * 2);
  bf16_t* WinT   = (bf16_t*)take((size_t)DINPROJ * DMODEL * 2);
  bf16_t* WoutT  = (bf16_t*)take((size_t)DMODEL * DINNER * 2);
  float*  dtv    = (float*)take((size_t)ROWS * NHEADS * 4);
  float*  adt    = (float*)take((size_t)ROWS * NHEADS * 4);
  float*  xs     = (float*)take((size_t)ROWS * DINNER * 4);
  bf16_t* xdtT   = (bf16_t*)take((size_t)ROWS * DINNER * 2);
  bf16_t* Bb     = (bf16_t*)take((size_t)ROWS * DSTATE * 2);
  bf16_t* Btb    = (bf16_t*)take((size_t)ROWS * DSTATE * 2);
  bf16_t* Cb     = (bf16_t*)take((size_t)ROWS * DSTATE * 2);
  float*  G      = (float*)take((size_t)BATCH * NCHUNK * CHUNK * CHUNK * 4);
  float*  chunkA = (float*)take((size_t)BATCH * NHEADS * NCHUNK * 4);
  float*  S      = (float*)take((size_t)BATCH * NCHUNK * NHEADS * HEADDIM * DSTATE * 4);
  float*  yb     = (float*)take((size_t)ROWS * DINNER * 4);
  bf16_t* u_bf   = (bf16_t*)take((size_t)ROWS * DINNER * 2);
  (void)ws_size; (void)n_in; (void)in_sizes; (void)out_size;

  // 1) casts / transposes
  {
    size_t n = (size_t)ROWS * DMODEL;
    k_cast_bf16<<<dim3((unsigned)((n + 255) / 256)), 256, 0, stream>>>(x, x_bf, n);
  }
  {
    size_t n = (size_t)DMODEL * DINPROJ;
    k_cast_transpose<<<dim3((unsigned)((n + 255) / 256)), 256, 0, stream>>>(
        W_in, WinT, DMODEL, DINPROJ);
  }
  {
    size_t n = (size_t)DINNER * DMODEL;
    k_cast_transpose<<<dim3((unsigned)((n + 255) / 256)), 256, 0, stream>>>(
        W_out, WoutT, DINNER, DMODEL);
  }

  // 2) in-projection GEMM: zx[8192,4384] (double-buffered TDM weights)
  k_gemm_bf16_tdm<<<dim3(DINPROJ / 32, ROWS / 512), 256, 0, stream>>>(
      x_bf, WinT, zx, ROWS, DINPROJ, DMODEL);

  // 3) dt softplus + A*dt
  k_dt<<<dim3((ROWS * NHEADS) / 256), 256, 0, stream>>>(zx, dt_bias, A_log, dtv, adt);

  // 4) conv + silu + channel routing
  k_conv<<<dim3(CONVDIM / 256, SEQ, BATCH), 256, 0, stream>>>(
      zx, conv_w, conv_b, dtv, xs, xdtT, Bb, Btb, Cb);

  // 5) per-chunk Gram G = C B^T
  k_gram<<<dim3(CHUNK / 64, CHUNK / 16, BATCH * NCHUNK), 32, 0, stream>>>(Cb, Bb, G);

  // 6) chunk-end states
  k_states<<<dim3(NCHUNK, NHEADS, BATCH), 256, 0, stream>>>(adt, xdtT, Btb, S, chunkA);

  // 7) inter-chunk recurrence (in place)
  k_scan<<<dim3(BATCH * NHEADS), 256, 0, stream>>>(S, chunkA);

  // 8) y = intra + inter + D skip
  k_y<<<dim3(NCHUNK, NHEADS, BATCH), 256, 0, stream>>>(adt, G, xdtT, Cb, S, xs, Dp, yb);

  // 9) gated RMSNorm
  k_rmsnorm<<<dim3(ROWS), 256, 0, stream>>>(yb, zx, norm_w, u_bf);

  // 10) out-projection GEMM -> d_out (double-buffered TDM weights)
  k_gemm_bf16_tdm<<<dim3(DMODEL / 32, ROWS / 512), 256, 0, stream>>>(
      u_bf, WoutT, out, ROWS, DMODEL, DINNER);
}
